// EMA_VQ_88381837017746
// MI455X (gfx1250) — matrix operand order repaired and checked by reading, not compiled, not run on hardware
//
#include <hip/hip_runtime.h>
#include <hip/hip_bf16.h>

// Problem shape (fixed by reference setup_inputs)
#define NROWS   32768
#define DIMH    512            // dim
#define D2      1024           // 2*dim
#define KCODES  8192
#define DECAYF  0.99f
#define ONE_M_D 0.01f

typedef __attribute__((ext_vector_type(16))) __bf16 v16bf;
typedef __attribute__((ext_vector_type(8)))  float  v8f;

union Frag { v16bf bf; uint4 q[2]; };

__device__ __forceinline__ unsigned short f32_to_bf16_rne(float x) {
    unsigned int u = __float_as_uint(x);
    u = (u + 0x7FFFu + ((u >> 16) & 1u)) >> 16;
    return (unsigned short)u;
}

// ---------------------------------------------------------------------------
// Zero accumulator region (n_total | dw | cs | ent_acc), contiguous floats.
// ---------------------------------------------------------------------------
__global__ void zero_kernel(float* __restrict__ p, unsigned long long n) {
    unsigned long long i = (unsigned long long)blockIdx.x * blockDim.x + threadIdx.x;
    unsigned long long stride = (unsigned long long)gridDim.x * blockDim.x;
    for (; i < n; i += stride) p[i] = 0.0f;
}

// ---------------------------------------------------------------------------
// Stream-convert z_flat (concat of z_real/z_imag) to bf16 in workspace.
// ---------------------------------------------------------------------------
__global__ void prep_z_kernel(const float* __restrict__ z_real,
                              const float* __restrict__ z_imag,
                              unsigned short* __restrict__ z_bf16) {
    unsigned long long i = (unsigned long long)blockIdx.x * blockDim.x + threadIdx.x;
    const unsigned long long total  = (unsigned long long)NROWS * D2;
    const unsigned long long stride = (unsigned long long)gridDim.x * blockDim.x;
    for (; i < total; i += stride) {
        unsigned long long r = i >> 10;
        int c = (int)(i & (D2 - 1));
        float x = (c < DIMH) ? z_real[r * DIMH + c]
                             : z_imag[r * DIMH + (c - DIMH)];
        z_bf16[i] = f32_to_bf16_rne(x);
    }
}

// ---------------------------------------------------------------------------
// Per-code: convert emb_w row to bf16, compute y_sq[k] = sum(e*e).
// One 256-thread block per code.
// ---------------------------------------------------------------------------
__global__ void prep_emb_kernel(const float* __restrict__ emb_w,
                                unsigned short* __restrict__ emb_bf16,
                                float* __restrict__ y_sq) {
    const int k = blockIdx.x;
    const int t = threadIdx.x;
    const float* row = emb_w + (size_t)k * D2;
    float ss = 0.0f;
    #pragma unroll
    for (int c = t; c < D2; c += 256) {
        float x = row[c];
        ss += x * x;
        emb_bf16[(size_t)k * D2 + c] = f32_to_bf16_rne(x);
    }
    #pragma unroll
    for (int off = 16; off > 0; off >>= 1) ss += __shfl_xor(ss, off, 32);
    __shared__ float wsum[8];
    if ((t & 31) == 0) wsum[t >> 5] = ss;
    __syncthreads();
    if (t == 0) {
        float s = 0.0f;
        #pragma unroll
        for (int i = 0; i < 8; ++i) s += wsum[i];
        y_sq[k] = s;
    }
}

// ---------------------------------------------------------------------------
// Fused distance-GEMM + argmin.
//  - 256 threads (8 waves), 128 z-rows per block, staged into 256 KB LDS
//    via async global->LDS copies (ASYNCcnt path).
//  - Each wave: its 16 rows x all K, FOUR independent column chains per
//    kt-group (64 codes), sharing one A fragment per dc step -> A LDS reads
//    amortized 4x and 4 independent WMMA chains for the matrix pipe.
//  - A LDS offset is laundered through asm volatile per kt-group so the
//    compiler re-reads A from LDS instead of hoisting ~256 VGPRs of A
//    (round-2 pathology), leaving room for double-buffered B.
// score(k) = y_sq[k] - 2 * <z_row, e_k>   (x_sq constant per row -> dropped)
// ---------------------------------------------------------------------------
__global__ __launch_bounds__(256)
void vq_argmin_kernel(const unsigned short* __restrict__ z_bf16,
                      const unsigned short* __restrict__ emb_bf16,
                      const float* __restrict__ y_sq,
                      int* __restrict__ idx_out) {
    extern __shared__ unsigned short zt[];          // [128][1024] bf16
    const int tid  = threadIdx.x;                   // 0..255
    const int wave = tid >> 5;                      // 0..7
    const int lane = tid & 31;
    const int row0 = blockIdx.x * 128;

    // ---- Stage 128 z rows via async global->LDS copies (16B chunks) ----
    {
        const unsigned short* zsrc = z_bf16 + (size_t)row0 * D2;
        unsigned lds_base = (unsigned)(uintptr_t)(void*)zt;
        for (int i = tid; i < (128 * D2) / 8; i += 256) {    // 8 bf16 = 16B
            unsigned long long ga =
                (unsigned long long)(uintptr_t)(zsrc + (size_t)i * 8);
            unsigned la = lds_base + (unsigned)i * 16u;
            asm volatile("global_load_async_to_lds_b128 %0, %1, off"
                         :: "v"(la), "v"(ga) : "memory");
        }
        asm volatile("s_wait_asynccnt 0x0" ::: "memory");
        __syncthreads();
    }

    const int wrow = wave * 16;       // wave's first LDS row
    const int m    = lane & 15;       // matrix row (A) / column (B,C) id
    const int g    = lane >> 4;       // lane-group (K-half selector)

    float best[8];
    int   bidx[8];
    #pragma unroll
    for (int j = 0; j < 8; ++j) { best[j] = 3.4e38f; bidx[j] = 0; }

    const unsigned aoff0 = (unsigned)((wrow + m) * D2 + 16 * g);

    for (int ktg = 0; ktg < KCODES / 64; ++ktg) {
        // Opaque per-iteration A offset: blocks LICM of the A LDS loads.
        unsigned aoff = aoff0;
        asm volatile("" : "+v"(aoff));
        const unsigned short* arow = zt + aoff;

        // Four B columns: codes ktg*64 + {0,16,32,48} + m
        const unsigned short* b0 =
            emb_bf16 + (size_t)(ktg * 64 + m) * D2 + g * 16;
        const unsigned short* b1 = b0 + (size_t)16 * D2;
        const unsigned short* b2 = b0 + (size_t)32 * D2;
        const unsigned short* b3 = b0 + (size_t)48 * D2;

        v8f acc0 = {}, acc1 = {}, acc2 = {}, acc3 = {};

        // Prologue: load step 0
        Frag A, B0, B1, B2, B3;
        A.q[0]  = *(const uint4*)(arow);
        A.q[1]  = *(const uint4*)(arow + 8);
        B0.q[0] = *(const uint4*)(b0);  B0.q[1] = *(const uint4*)(b0 + 8);
        B1.q[0] = *(const uint4*)(b1);  B1.q[1] = *(const uint4*)(b1 + 8);
        B2.q[0] = *(const uint4*)(b2);  B2.q[1] = *(const uint4*)(b2 + 8);
        B3.q[0] = *(const uint4*)(b3);  B3.q[1] = *(const uint4*)(b3 + 8);

        // Pipelined main loop: prefetch dc+32 while computing dc
        #pragma unroll 4
        for (int dc = 0; dc < D2 - 32; dc += 32) {
            Frag An, B0n, B1n, B2n, B3n;
            const int dn = dc + 32;
            An.q[0]  = *(const uint4*)(arow + dn);
            An.q[1]  = *(const uint4*)(arow + dn + 8);
            B0n.q[0] = *(const uint4*)(b0 + dn);
            B0n.q[1] = *(const uint4*)(b0 + dn + 8);
            B1n.q[0] = *(const uint4*)(b1 + dn);
            B1n.q[1] = *(const uint4*)(b1 + dn + 8);
            B2n.q[0] = *(const uint4*)(b2 + dn);
            B2n.q[1] = *(const uint4*)(b2 + dn + 8);
            B3n.q[0] = *(const uint4*)(b3 + dn);
            B3n.q[1] = *(const uint4*)(b3 + dn + 8);

            acc0 = __builtin_amdgcn_wmma_f32_16x16x32_bf16(
                false, A.bf, false, B0.bf, (short)0, acc0, false, false);
            acc1 = __builtin_amdgcn_wmma_f32_16x16x32_bf16(
                false, A.bf, false, B1.bf, (short)0, acc1, false, false);
            acc2 = __builtin_amdgcn_wmma_f32_16x16x32_bf16(
                false, A.bf, false, B2.bf, (short)0, acc2, false, false);
            acc3 = __builtin_amdgcn_wmma_f32_16x16x32_bf16(
                false, A.bf, false, B3.bf, (short)0, acc3, false, false);

            A = An; B0 = B0n; B1 = B1n; B2 = B2n; B3 = B3n;
        }
        // Epilogue step
        acc0 = __builtin_amdgcn_wmma_f32_16x16x32_bf16(
            false, A.bf, false, B0.bf, (short)0, acc0, false, false);
        acc1 = __builtin_amdgcn_wmma_f32_16x16x32_bf16(
            false, A.bf, false, B1.bf, (short)0, acc1, false, false);
        acc2 = __builtin_amdgcn_wmma_f32_16x16x32_bf16(
            false, A.bf, false, B2.bf, (short)0, acc2, false, false);
        acc3 = __builtin_amdgcn_wmma_f32_16x16x32_bf16(
            false, A.bf, false, B3.bf, (short)0, acc3, false, false);

        // Score + running argmin for this lane's four columns.
        const int   col0 = ktg * 64 + m;
        const float ys0  = y_sq[col0];
        const float ys1  = y_sq[col0 + 16];
        const float ys2  = y_sq[col0 + 32];
        const float ys3  = y_sq[col0 + 48];
        #pragma unroll
        for (int j = 0; j < 8; ++j) {           // VGPR j: row (g*8 + j)
            float s0 = ys0 - 2.0f * acc0[j];
            float s1 = ys1 - 2.0f * acc1[j];
            float s2 = ys2 - 2.0f * acc2[j];
            float s3 = ys3 - 2.0f * acc3[j];
            if (s0 < best[j]) { best[j] = s0; bidx[j] = col0; }
            if (s1 < best[j]) { best[j] = s1; bidx[j] = col0 + 16; }
            if (s2 < best[j]) { best[j] = s2; bidx[j] = col0 + 32; }
            if (s3 < best[j]) { best[j] = s3; bidx[j] = col0 + 48; }
        }
    }

    // Min-reduce (value, index) across the 16 lanes of each half-wave.
    #pragma unroll
    for (int off = 1; off < 16; off <<= 1) {
        #pragma unroll
        for (int j = 0; j < 8; ++j) {
            float ov = __shfl_xor(best[j], off, 32);
            int   oi = __shfl_xor(bidx[j], off, 32);
            if (ov < best[j] || (ov == best[j] && oi < bidx[j])) {
                best[j] = ov; bidx[j] = oi;
            }
        }
    }
    if (m == 0) {                               // lanes 0 and 16 hold results
        const int rbase = row0 + wrow + g * 8;
        #pragma unroll
        for (int j = 0; j < 8; ++j) idx_out[rbase + j] = bidx[j];
    }
}

// ---------------------------------------------------------------------------
// Per-row: gather z_q = emb_w[idx], write z_q_r/z_q_i, loss, indices(out),
// scatter-add n_total and dw.  One 256-thread block per row.
// ---------------------------------------------------------------------------
__global__ void gather_scatter_kernel(const float* __restrict__ z_real,
                                      const float* __restrict__ z_imag,
                                      const float* __restrict__ emb_w,
                                      const int* __restrict__ idx,
                                      float* __restrict__ o_zqr,
                                      float* __restrict__ o_zqi,
                                      float* __restrict__ o_loss,
                                      float* __restrict__ o_idx,
                                      float* __restrict__ n_total,
                                      float* __restrict__ dw) {
    const int n = blockIdx.x;
    const int t = threadIdx.x;
    const int k = idx[n];
    const float* erow = emb_w + (size_t)k * D2;
    float l = 0.0f;
    #pragma unroll
    for (int c = t; c < D2; c += 256) {
        float e = erow[c];
        float z;
        if (c < DIMH) {
            z = z_real[(size_t)n * DIMH + c];
            o_zqr[(size_t)n * DIMH + c] = e;
        } else {
            z = z_imag[(size_t)n * DIMH + (c - DIMH)];
            o_zqi[(size_t)n * DIMH + (c - DIMH)] = e;
        }
        float d = e - z;
        l += d * d;
        atomicAdd(&dw[(size_t)k * D2 + c], z);
    }
    #pragma unroll
    for (int off = 16; off > 0; off >>= 1) l += __shfl_xor(l, off, 32);
    __shared__ float wsum[8];
    if ((t & 31) == 0) wsum[t >> 5] = l;
    __syncthreads();
    if (t == 0) {
        float s = 0.0f;
        #pragma unroll
        for (int i = 0; i < 8; ++i) s += wsum[i];
        o_loss[n] = 0.25f * s * (1.0f / (float)D2);   // BETA * mean
        o_idx[n]  = (float)k;
        atomicAdd(&n_total[k], 1.0f);
    }
}

// ---------------------------------------------------------------------------
// Per-code cluster-size EMA + entropy accumulation.
// ---------------------------------------------------------------------------
__global__ void cluster_kernel(const float* __restrict__ ema_cs,
                               const float* __restrict__ n_total,
                               float* __restrict__ o_new_cs,
                               float* __restrict__ cs_ws,
                               float* __restrict__ ent_acc) {
    const int k = blockIdx.x * blockDim.x + threadIdx.x;
    if (k >= KCODES) return;
    float nt  = n_total[k];
    float ncs = ema_cs[k] * DECAYF + ONE_M_D * nt;
    o_new_cs[k] = ncs;
    float cs = ncs + 1e-5f;
    if (cs < 1.0f) cs += 1.0f;
    cs_ws[k] = cs;
    float p = nt * (1.0f / (float)NROWS);
    atomicAdd(ent_acc, p * __logf(p + 1e-10f));
}

__global__ void entropy_final_kernel(const float* __restrict__ ent_acc,
                                     float* __restrict__ o_ent) {
    if (threadIdx.x == 0 && blockIdx.x == 0)
        o_ent[0] = -ent_acc[0] / __logf((float)KCODES);
}

// ---------------------------------------------------------------------------
// EMA codebook update: new_ema_w, new_w (clip(new_ema_w / cs, -5, 5)).
// ---------------------------------------------------------------------------
__global__ void ema_update_kernel(const float* __restrict__ ema_w,
                                  const float* __restrict__ dw,
                                  const float* __restrict__ cs,
                                  float* __restrict__ o_new_ema_w,
                                  float* __restrict__ o_new_w) {
    unsigned long long i = (unsigned long long)blockIdx.x * blockDim.x + threadIdx.x;
    const unsigned long long total  = (unsigned long long)KCODES * D2;
    const unsigned long long stride = (unsigned long long)gridDim.x * blockDim.x;
    for (; i < total; i += stride) {
        float w = ema_w[i] * DECAYF + ONE_M_D * dw[i];
        o_new_ema_w[i] = w;
        float nw = w / cs[i >> 10];
        nw = fminf(5.0f, fmaxf(-5.0f, nw));
        o_new_w[i] = nw;
    }
}

// ---------------------------------------------------------------------------
extern "C" void kernel_launch(void* const* d_in, const int* in_sizes, int n_in,
                              void* d_out, int out_size, void* d_ws, size_t ws_size,
                              hipStream_t stream) {
    (void)in_sizes; (void)n_in; (void)out_size; (void)ws_size;

    const float* z_real = (const float*)d_in[0];
    const float* z_imag = (const float*)d_in[1];
    const float* emb_w  = (const float*)d_in[2];
    const float* ema_cs = (const float*)d_in[3];
    const float* ema_w  = (const float*)d_in[4];

    // Output layout (floats, reference return order)
    float* out      = (float*)d_out;
    float* o_zqr    = out;                                   // N*DIMH
    float* o_zqi    = o_zqr + (size_t)NROWS * DIMH;          // N*DIMH
    float* o_loss   = o_zqi + (size_t)NROWS * DIMH;          // N
    float* o_idx    = o_loss + NROWS;                        // N
    float* o_ent    = o_idx + NROWS;                         // 1
    float* o_neww   = o_ent + 1;                             // K*D2
    float* o_ncs    = o_neww + (size_t)KCODES * D2;          // K
    float* o_nemaw  = o_ncs + KCODES;                        // K*D2

    // Workspace layout
    char* ws = (char*)d_ws;
    unsigned short* emb_bf16 = (unsigned short*)ws;                    // K*D2 bf16
    unsigned short* z_bf16   = emb_bf16 + (size_t)KCODES * D2;         // N*D2 bf16
    float* y_sq    = (float*)(z_bf16 + (size_t)NROWS * D2);            // K
    int*   idx     = (int*)((char*)y_sq + (size_t)KCODES * 4);         // N
    float* n_total = (float*)((char*)idx + (size_t)NROWS * 4);         // K   (zeroed)
    float* dw      = n_total + KCODES;                                 // K*D2 (zeroed)
    float* cs_ws   = dw + (size_t)KCODES * D2;                         // K   (zeroed)
    float* ent_acc = cs_ws + KCODES;                                   // 1   (zeroed)

    const unsigned long long zero_n =
        (unsigned long long)KCODES + (unsigned long long)KCODES * D2 + KCODES + 1;

    zero_kernel<<<4096, 256, 0, stream>>>(n_total, zero_n);

    prep_z_kernel<<<8192, 256, 0, stream>>>(z_real, z_imag, z_bf16);

    prep_emb_kernel<<<KCODES, 256, 0, stream>>>(emb_w, emb_bf16, y_sq);

    vq_argmin_kernel<<<NROWS / 128, 256, 128 * D2 * sizeof(unsigned short), stream>>>(
        z_bf16, emb_bf16, y_sq, idx);

    gather_scatter_kernel<<<NROWS, 256, 0, stream>>>(
        z_real, z_imag, emb_w, idx, o_zqr, o_zqi, o_loss, o_idx, n_total, dw);

    cluster_kernel<<<(KCODES + 255) / 256, 256, 0, stream>>>(
        ema_cs, n_total, o_ncs, cs_ws, ent_acc);

    entropy_final_kernel<<<1, 32, 0, stream>>>(ent_acc, o_ent);

    ema_update_kernel<<<4096, 256, 0, stream>>>(ema_w, dw, cs_ws, o_nemaw, o_neww);
}
